// SparseMoEBlock_9328668967093
// MI455X (gfx1250) — compile-verified, matrix operand order checked by reading, not run on hardware
//
#include <hip/hip_runtime.h>

// ---------------- types ----------------
typedef __bf16 bf16_t;
typedef __attribute__((ext_vector_type(16))) __bf16 v16bf;
typedef __attribute__((ext_vector_type(8)))  __bf16 v8bf;
typedef __attribute__((ext_vector_type(8)))  float  v8f;

// ---------------- problem sizes ----------------
#define DD   1152
#define EE   8
#define SS   16384
#define KK   4096
#define HEXP 1152
#define HSH  4608

// ---------------- scalar helpers ----------------
__device__ __forceinline__ bf16_t f2bf(float f) {
  unsigned u = __builtin_bit_cast(unsigned, f);
  u += 0x7FFFu + ((u >> 16) & 1u);               // round-to-nearest-even
  unsigned short h = (unsigned short)(u >> 16);
  return __builtin_bit_cast(bf16_t, h);
}
__device__ __forceinline__ float bf2f(bf16_t b) {
  unsigned short h = __builtin_bit_cast(unsigned short, b);
  unsigned u = ((unsigned)h) << 16;
  return __builtin_bit_cast(float, u);
}
__device__ __forceinline__ float gelu_tanh(float v) {
  float v3 = v * v * v;
  return 0.5f * v * (1.0f + tanhf(0.7978845608028654f * (v + 0.044715f * v3)));
}
__device__ __forceinline__ float silu_f(float v) {
  return v / (1.0f + __expf(-v));
}

// ---------------- CDNA5 async global->LDS DMA (ASYNCcnt) ----------------
__device__ __forceinline__ void async_b128(unsigned lds_off, const void* gaddr) {
  unsigned long long ga = (unsigned long long)(size_t)gaddr;
  asm volatile("global_load_async_to_lds_b128 %0, %1, off"
               :: "v"(lds_off), "v"(ga) : "memory");
}
__device__ __forceinline__ void wait_async0() {
  asm volatile("s_wait_asynccnt 0x0" ::: "memory");
}

// ---------------- zero init ----------------
__global__ void zero_kernel(float* __restrict__ p, long long n) {
  long long i = (long long)blockIdx.x * blockDim.x + threadIdx.x;
  long long stride = (long long)gridDim.x * blockDim.x;
  for (; i < n; i += stride) p[i] = 0.0f;
}

// ---------------- fp32 -> bf16 bulk convert ----------------
__global__ __launch_bounds__(256) void f32_to_bf16(
    const float* __restrict__ s, void* __restrict__ d, long long n) {
  bf16_t* dp = (bf16_t*)d;
  long long i = (long long)blockIdx.x * blockDim.x + threadIdx.x;
  long long stride = (long long)gridDim.x * blockDim.x;
  for (; i < n; i += stride) dp[i] = f2bf(s[i]);
}

// ---------------- fp32 [R][C] -> bf16 transposed [C][R] ----------------
__global__ __launch_bounds__(256) void transp_bf16(
    const float* __restrict__ s, void* __restrict__ d, int R, int C,
    unsigned long long sBatch, unsigned long long dBatch) {
  __shared__ bf16_t t[32][33];
  const int tx = threadIdx.x & 31;
  const int ty = threadIdx.x >> 5;          // 0..7
  const int c0 = blockIdx.x * 32;
  const int r0 = blockIdx.y * 32;
  const int bz = blockIdx.z;
  const float* sp = s + sBatch * bz;
  bf16_t* dp = (bf16_t*)d + dBatch * bz;
#pragma unroll
  for (int i = 0; i < 4; i++) {
    int r = r0 + ty + i * 8;
    t[ty + i * 8][tx] = f2bf(sp[(size_t)r * C + c0 + tx]);
  }
  __syncthreads();
#pragma unroll
  for (int i = 0; i < 4; i++) {
    int c = c0 + ty + i * 8;
    dp[(size_t)c * R + r0 + tx] = t[tx][ty + i * 8];
  }
}

// ---------------- double-buffered async WMMA GEMM (all-bf16 operands) ----
// A: [Mrows][lda] bf16 (optionally row-gathered via rowidx)
// Bt: [N][Kdim] bf16 (pre-transposed weights)
// EPI: 0 = gelu -> bf16, 1 = silu -> bf16,
//      2 = gated scatter atomicAdd fp32, 3 = atomicAdd fp32
template <bool GATHER, int EPI>
__global__ __launch_bounds__(256) void wmma_gemm_async(
    const void* __restrict__ Ap, unsigned long long Abatch, int lda,
    const void* __restrict__ Btp, unsigned long long Bbatch,
    const float* __restrict__ bias, int biasBatch,
    const int* __restrict__ rowidx, const float* __restrict__ gating,
    void* __restrict__ Out, int Mrows, int N, int Kdim) {
  constexpr unsigned BUFB = 128 * 40 * 2;   // bytes per LDS buffer
  __shared__ __align__(16) bf16_t ldsA[2][128 * 40];
  __shared__ __align__(16) bf16_t ldsB[2][128 * 40];

  const int tid = threadIdx.x;
  const int bz = blockIdx.z;
  const int bm = blockIdx.y;
  const int bn = blockIdx.x;

  const bf16_t* A = (const bf16_t*)Ap;
  const bf16_t* Bt = (const bf16_t*)Btp + (unsigned long long)bz * Bbatch;
  const float* biasb = bias + (size_t)bz * biasBatch;

  // each thread owns two (row, 16B-chunk) transfers per tile for A and B
  const int ra0 = tid >> 2;                  // 0..63
  const int ra1 = (tid + 256) >> 2;          // 64..127
  const int ca0 = tid & 3;
  const int ca1 = (tid + 256) & 3;           // == ca0, kept for clarity

  const bf16_t *ap0, *ap1;
  if (GATHER) {
    ap0 = A + (size_t)rowidx[(size_t)bz * Mrows + bm * 128 + ra0] * lda;
    ap1 = A + (size_t)rowidx[(size_t)bz * Mrows + bm * 128 + ra1] * lda;
  } else {
    ap0 = A + (unsigned long long)bz * Abatch + (size_t)(bm * 128 + ra0) * lda;
    ap1 = A + (unsigned long long)bz * Abatch + (size_t)(bm * 128 + ra1) * lda;
  }
  const bf16_t* bp0 = Bt + (size_t)(bn * 128 + ra0) * Kdim;
  const bf16_t* bp1 = Bt + (size_t)(bn * 128 + ra1) * Kdim;

  const unsigned ldsAbase = (unsigned)(size_t)(&ldsA[0][0]);
  const unsigned ldsBbase = (unsigned)(size_t)(&ldsB[0][0]);
  const unsigned offA0 = (unsigned)(ra0 * 40 + ca0 * 8) * 2;
  const unsigned offA1 = (unsigned)(ra1 * 40 + ca1 * 8) * 2;

  auto issue = [&](int k0, int buf) {
    unsigned la = ldsAbase + (unsigned)buf * BUFB;
    unsigned lb = ldsBbase + (unsigned)buf * BUFB;
    async_b128(la + offA0, ap0 + k0 + ca0 * 8);
    async_b128(la + offA1, ap1 + k0 + ca1 * 8);
    async_b128(lb + offA0, bp0 + k0 + ca0 * 8);
    async_b128(lb + offA1, bp1 + k0 + ca1 * 8);
  };

  const int lane = tid & 31;
  const int wid  = tid >> 5;
  const int wm = wid >> 1;  // 0..3 -> 32 rows each
  const int wn = wid & 1;   // 0..1 -> 64 cols each
  const int hi = lane >> 4;
  const int l15 = lane & 15;

  v8f acc[2][4];
#pragma unroll
  for (int i = 0; i < 2; i++)
#pragma unroll
    for (int j = 0; j < 4; j++)
#pragma unroll
      for (int r = 0; r < 8; r++) acc[i][j][r] = 0.0f;

  const int nk = Kdim / 32;
  issue(0, 0);
  for (int t = 0; t < nk; ++t) {
    const int buf = t & 1;
    wait_async0();           // this wave's DMA for tile t done
    __syncthreads();         // collectively: buffer `buf` fully written, prior reads done
    if (t + 1 < nk) issue((t + 1) * 32, buf ^ 1);

    const bf16_t* cA = &ldsA[buf][0];
    const bf16_t* cB = &ldsB[buf][0];
    // fragments: lanes 0-15 get K 0-7 & 16-23, lanes 16-31 get K 8-15 & 24-31
    v16bf afr[2];
    v16bf bfr[4];
#pragma unroll
    for (int i = 0; i < 2; i++) {
      const bf16_t* p = cA + (wm * 32 + i * 16 + l15) * 40 + hi * 8;
      v8bf lo = *(const v8bf*)p;
      v8bf hi8 = *(const v8bf*)(p + 16);
#pragma unroll
      for (int q = 0; q < 8; q++) { afr[i][q] = lo[q]; afr[i][q + 8] = hi8[q]; }
    }
#pragma unroll
    for (int j = 0; j < 4; j++) {
      const bf16_t* p = cB + (wn * 64 + j * 16 + l15) * 40 + hi * 8;
      v8bf lo = *(const v8bf*)p;
      v8bf hi8 = *(const v8bf*)(p + 16);
#pragma unroll
      for (int q = 0; q < 8; q++) { bfr[j][q] = lo[q]; bfr[j][q + 8] = hi8[q]; }
    }
#pragma unroll
    for (int i = 0; i < 2; i++)
#pragma unroll
      for (int j = 0; j < 4; j++)
        acc[i][j] = __builtin_amdgcn_wmma_f32_16x16x32_bf16(
            false, afr[i], false, bfr[j], (short)0, acc[i][j], false, false);
  }

  // epilogue: vgpr r holds M=r (lanes 0-15) / M=8+r (lanes 16-31), N=lane&15
#pragma unroll
  for (int i = 0; i < 2; i++) {
#pragma unroll
    for (int j = 0; j < 4; j++) {
      int gn = bn * 128 + wn * 64 + j * 16 + l15;
      float bv = biasb[gn];
#pragma unroll
      for (int r = 0; r < 8; r++) {
        int Mi = bm * 128 + wm * 32 + i * 16 + r + hi * 8;
        float val = acc[i][j][r] + bv;
        if (EPI == 0) {
          ((bf16_t*)Out)[((size_t)bz * Mrows + Mi) * (size_t)N + gn] = f2bf(gelu_tanh(val));
        } else if (EPI == 1) {
          ((bf16_t*)Out)[((size_t)bz * Mrows + Mi) * (size_t)N + gn] = f2bf(silu_f(val));
        } else if (EPI == 2) {
          int tok = rowidx[(size_t)bz * Mrows + Mi];
          float g = gating[(size_t)bz * Mrows + Mi];
          atomicAdd((float*)Out + (size_t)tok * N + gn, g * val);
        } else {
          atomicAdd((float*)Out + (size_t)Mi * N + gn, val);
        }
      }
    }
  }
}

// ---------------- router: logits -> softmax -> scores[E,S] ----------------
__global__ __launch_bounds__(128) void router_softmax(
    const float* __restrict__ x, const float* __restrict__ gw,
    float* __restrict__ scores) {
  const int t = blockIdx.x;
  const int tid = threadIdx.x;
  float a[EE];
#pragma unroll
  for (int e = 0; e < EE; e++) a[e] = 0.0f;
  const float* xr = x + (size_t)t * DD;
  for (int d = tid; d < DD; d += 128) {
    float xv = xr[d];
#pragma unroll
    for (int e = 0; e < EE; e++) a[e] += xv * gw[e * DD + d];
  }
  __shared__ float red[EE][128];
#pragma unroll
  for (int e = 0; e < EE; e++) red[e][tid] = a[e];
  __syncthreads();
  for (int s = 64; s > 0; s >>= 1) {
    if (tid < s) {
#pragma unroll
      for (int e = 0; e < EE; e++) red[e][tid] += red[e][tid + s];
    }
    __syncthreads();
  }
  if (tid == 0) {
    float m = red[0][0];
#pragma unroll
    for (int e = 1; e < EE; e++) m = fmaxf(m, red[e][0]);
    float p[EE], sum = 0.0f;
#pragma unroll
    for (int e = 0; e < EE; e++) { p[e] = __expf(red[e][0] - m); sum += p[e]; }
    float inv = 1.0f / sum;
#pragma unroll
    for (int e = 0; e < EE; e++) scores[(size_t)e * SS + t] = p[e] * inv;
  }
}

// ---------------- top-K per expert via bisection in LDS ----------------
__global__ __launch_bounds__(1024) void topk_kernel(
    const float* __restrict__ scores, int* __restrict__ idx,
    float* __restrict__ gate, float* __restrict__ ones) {
  __shared__ float sc[SS];   // 64 KB: whole expert row (320 KB WGP LDS)
  __shared__ int cnt;
  const int e = blockIdx.x;
  const int tid = threadIdx.x;
  const float* s = scores + (size_t)e * SS;
  for (int i = tid; i < SS; i += 1024) sc[i] = s[i];
  __syncthreads();

  // invariant: count(>lo) >= K, count(>hi) < K   (softmax probs in (0,1))
  float lo = 0.0f, hi = 1.0f;
  for (int it = 0; it < 32; ++it) {
    float mid = 0.5f * (lo + hi);
    if (tid == 0) cnt = 0;
    __syncthreads();
    int c = 0;
    for (int i = tid; i < SS; i += 1024) c += (sc[i] > mid) ? 1 : 0;
    atomicAdd(&cnt, c);
    __syncthreads();
    int total = cnt;
    __syncthreads();
    if (total >= KK) lo = mid; else hi = mid;
  }
  if (tid == 0) cnt = 0;
  __syncthreads();
  for (int i = tid; i < SS; i += 1024) {
    float v = sc[i];
    if (v > hi) {
      int p = atomicAdd(&cnt, 1);
      idx[(size_t)e * KK + p] = i;
      gate[(size_t)e * KK + p] = v;
    }
  }
  __syncthreads();
  for (int i = tid; i < SS; i += 1024) {
    float v = sc[i];
    if (v <= hi && v > lo) {
      int p = atomicAdd(&cnt, 1);
      if (p < KK) {
        idx[(size_t)e * KK + p] = i;
        gate[(size_t)e * KK + p] = v;
      }
    }
  }
  __syncthreads();
  for (int j = tid; j < KK; j += 1024) {
    int t = idx[(size_t)e * KK + j];
    ones[(size_t)t * EE + e] = 1.0f;
  }
}

// ---------------- capacity fc2: [S,D](bf16) @ [D,8] + b ----------------
__global__ __launch_bounds__(128) void cap_out_kernel(
    const void* __restrict__ h1v, const float* __restrict__ w2,
    const float* __restrict__ b2, float* __restrict__ out) {
  const bf16_t* h1 = (const bf16_t*)h1v;
  const int t = blockIdx.x;
  const int tid = threadIdx.x;
  float a[EE];
#pragma unroll
  for (int e = 0; e < EE; e++) a[e] = 0.0f;
  const bf16_t* hr = h1 + (size_t)t * DD;
  for (int d = tid; d < DD; d += 128) {
    float h = bf2f(hr[d]);
#pragma unroll
    for (int e = 0; e < EE; e++) a[e] += h * w2[d * EE + e];
  }
  __shared__ float red[EE][128];
#pragma unroll
  for (int e = 0; e < EE; e++) red[e][tid] = a[e];
  __syncthreads();
  for (int s = 64; s > 0; s >>= 1) {
    if (tid < s) {
#pragma unroll
      for (int e = 0; e < EE; e++) red[e][tid] += red[e][tid + s];
    }
    __syncthreads();
  }
  if (tid == 0) {
#pragma unroll
    for (int e = 0; e < EE; e++) out[(size_t)t * EE + e] = red[e][0] + b2[e];
  }
}

// ---------------- launcher ----------------
extern "C" void kernel_launch(void* const* d_in, const int* in_sizes, int n_in,
                              void* d_out, int out_size, void* d_ws, size_t ws_size,
                              hipStream_t stream) {
  (void)in_sizes; (void)n_in; (void)out_size; (void)ws_size;
  const float* x      = (const float*)d_in[0];
  const float* gw     = (const float*)d_in[1];
  const float* cap_w1 = (const float*)d_in[2];
  const float* cap_b1 = (const float*)d_in[3];
  const float* cap_w2 = (const float*)d_in[4];
  const float* cap_b2 = (const float*)d_in[5];
  const float* exp_w1 = (const float*)d_in[6];
  const float* exp_b1 = (const float*)d_in[7];
  const float* exp_w2 = (const float*)d_in[8];
  const float* exp_b2 = (const float*)d_in[9];
  const float* sh_w1  = (const float*)d_in[10];
  const float* sh_b1  = (const float*)d_in[11];
  const float* sh_w2  = (const float*)d_in[12];
  const float* sh_b2  = (const float*)d_in[13];

  float* xout = (float*)d_out;                       // [S,D]
  float* ones = xout + (size_t)SS * DD;              // [S,E]
  float* capo = ones + (size_t)SS * EE;              // [S,E]

  char* ws = (char*)d_ws;
  size_t off = 0;
  auto take = [&](size_t sz) {
    void* p = (void*)(ws + off);
    off += (sz + 255) & ~(size_t)255;
    return p;
  };
  void* xb    = take((size_t)SS * DD * 2);           // bf16 x
  void* wcap1 = take((size_t)DD * DD * 2);           // bf16 [D][D] transposed
  void* wsh1  = take((size_t)HSH * DD * 2);          // bf16 [HSH][D]
  void* wsh2  = take((size_t)DD * HSH * 2);          // bf16 [D][HSH]
  void* we1   = take((size_t)EE * HEXP * DD * 2);    // bf16 [E][HEXP][D]
  void* we2   = take((size_t)EE * DD * HEXP * 2);    // bf16 [E][D][HEXP]
  void* h_sh  = take((size_t)SS * HSH * 2);          // bf16 shared hidden
  void* h1    = h_sh;                                // bf16 [S,D] (dead before h_sh written)
  void* h_exp = take((size_t)EE * KK * HEXP * 2);    // bf16 expert hidden
  float* scores = (float*)take((size_t)EE * SS * 4);
  int*   idx    = (int*)take((size_t)EE * KK * 4);
  float* gate   = (float*)take((size_t)EE * KK * 4);

  // 1. zero x_out + ones sections
  long long nz = (long long)SS * DD + (long long)SS * EE;
  zero_kernel<<<dim3(4096), dim3(256), 0, stream>>>(xout, nz);

  // 2. one-time bf16 conversions / weight transposes
  f32_to_bf16<<<dim3(4096), dim3(256), 0, stream>>>(x, xb, (long long)SS * DD);
  transp_bf16<<<dim3(DD / 32, DD / 32, 1), dim3(256), 0, stream>>>(
      cap_w1, wcap1, DD, DD, 0ull, 0ull);
  transp_bf16<<<dim3(HSH / 32, DD / 32, 1), dim3(256), 0, stream>>>(
      sh_w1, wsh1, DD, HSH, 0ull, 0ull);
  transp_bf16<<<dim3(DD / 32, HSH / 32, 1), dim3(256), 0, stream>>>(
      sh_w2, wsh2, HSH, DD, 0ull, 0ull);
  transp_bf16<<<dim3(HEXP / 32, DD / 32, EE), dim3(256), 0, stream>>>(
      exp_w1, we1, DD, HEXP, (unsigned long long)DD * HEXP,
      (unsigned long long)HEXP * DD);
  transp_bf16<<<dim3(DD / 32, HEXP / 32, EE), dim3(256), 0, stream>>>(
      exp_w2, we2, HEXP, DD, (unsigned long long)HEXP * DD,
      (unsigned long long)DD * HEXP);

  // 3. capacity hidden: h1 = silu(x @ cap_w1 + b1) -> bf16
  wmma_gemm_async<false, 1><<<dim3(DD / 128, SS / 128, 1), dim3(256), 0, stream>>>(
      xb, 0ull, DD, wcap1, 0ull, cap_b1, 0, nullptr, nullptr, h1, SS, DD, DD);

  // 4. capacity logits
  cap_out_kernel<<<dim3(SS), dim3(128), 0, stream>>>(h1, cap_w2, cap_b2, capo);

  // 5. router softmax -> scores[E,S]
  router_softmax<<<dim3(SS), dim3(128), 0, stream>>>(x, gw, scores);

  // 6. expert-major top-K (+ routing mask)
  topk_kernel<<<dim3(EE), dim3(1024), 0, stream>>>(scores, idx, gate, ones);

  // 7. shared fc1: h_sh = gelu(x @ sh_w1 + b1) -> bf16  (overwrites h1 region)
  wmma_gemm_async<false, 0><<<dim3(HSH / 128, SS / 128, 1), dim3(256), 0, stream>>>(
      xb, 0ull, DD, wsh1, 0ull, sh_b1, 0, nullptr, nullptr, h_sh, SS, HSH, DD);

  // 8. expert fc1 (gather rows through idx)
  wmma_gemm_async<true, 0><<<dim3(HEXP / 128, KK / 128, EE), dim3(256), 0, stream>>>(
      xb, 0ull, DD, we1, (unsigned long long)HEXP * DD, exp_b1, HEXP,
      idx, nullptr, h_exp, KK, HEXP, DD);

  // 9. expert fc2 + gated scatter-add into x_out
  wmma_gemm_async<false, 2><<<dim3(DD / 128, KK / 128, EE), dim3(256), 0, stream>>>(
      h_exp, (unsigned long long)KK * HEXP, HEXP, we2,
      (unsigned long long)DD * HEXP, exp_b2, DD, idx, gate, xout, KK, DD, HEXP);

  // 10. shared fc2 + atomic add into x_out
  wmma_gemm_async<false, 3><<<dim3(DD / 128, SS / 128, 1), dim3(256), 0, stream>>>(
      h_sh, 0ull, HSH, wsh2, 0ull, sh_b2, 0, nullptr, nullptr, xout, SS, DD, HSH);
}
